// EquivariantTransformerBlock_34471407517994
// MI455X (gfx1250) — matrix operand order repaired and checked by reading, not compiled, not run on hardware
//
#include <hip/hip_runtime.h>
#include <hip/hip_bf16.h>
#include <math.h>

// Problem constants (from reference)
#define NN   10000          // nodes
#define EE   320000         // edges
#define CC   128            // channels
#define HH   8              // heads
#define DHD  16             // head dim
#define NPP  4              // points
#define HD   128            // H*DH
#define HP   32             // H*NP
#define C2   256            // 2*C
#define NV   30000          // N*3 (vector rows)

typedef __attribute__((ext_vector_type(16))) __bf16 v16bf;
typedef __attribute__((ext_vector_type(8)))  float  v8f;

// ---- ordered float <-> uint key for atomic segment-max ----
__device__ __forceinline__ unsigned fkey(float f) {
  unsigned u = __float_as_uint(f);
  return (u & 0x80000000u) ? ~u : (u | 0x80000000u);
}
__device__ __forceinline__ float keyf(unsigned k) {
  unsigned u = (k & 0x80000000u) ? (k & 0x7FFFFFFFu) : ~k;
  return __uint_as_float(u);
}

// =====================================================================
// Block-tiled WMMA GEMM machinery.
//  - B panel (K x NCOL fp32, ld ldb, col offset col0) is staged to LDS as
//    bf16, pre-swizzled into WMMA B-fragment order so each lane reads its
//    16 elements contiguously (2x ds_load_b128).
//  - Fragment (ct, ks) occupies 32 lanes * 16 bf16; lane L holds column
//    ct*16 + L%16, K-range ks*32 + (L/16)*16 + e  (contiguous k order, same as A).
//  - A fragments live in registers, loaded once per 16-row wave tile and
//    reused across all column tiles (k = ks*32 + (L/16)*16 + e, same order as B).
// =====================================================================

template<int K, int NCOL>
__device__ __forceinline__ void stage_B(const float* __restrict__ B, int ldb, int col0,
                                        __bf16* __restrict__ sB) {
  for (int idx = threadIdx.x; idx < K * NCOL; idx += (int)blockDim.x) {
    int k = idx / NCOL, col = idx - k * NCOL;
    int ct = col >> 4, lc = col & 15;
    int ks = k >> 5, k32 = k & 31;
    int laneidx = ((k32 >> 4) << 4) | lc;
    int e = k32 & 15;
    sB[(size_t)((ct * (K / 32) + ks) * 32 + laneidx) * 16 + e] =
        (__bf16)B[(long)k * ldb + col0 + col];
  }
}

// AMODE 0: row gr at A + gr*lda.  AMODE 1: x-layout (N,4,C) vector rows.
// SUBT: subtract tvec[gr] from every element of row gr (agg_p - t).
template<int K, int AMODE, bool SUBT>
__device__ __forceinline__ void load_A_frags(const float* __restrict__ A, int lda, int row0,
                                             const float* __restrict__ tvec,
                                             v16bf* __restrict__ af) {
  const int lane = threadIdx.x & 31;
  const int gr = row0 + (lane & 15);
  const float* Arow = (AMODE == 0) ? A + (long)gr * lda
                                   : A + (long)((gr / 3) * 4 + 1 + (gr % 3)) * lda;
  const float ts = SUBT ? tvec[gr] : 0.0f;
  const int ka = (lane >> 4) << 4;
#pragma unroll
  for (int ks = 0; ks < K / 32; ++ks) {
#pragma unroll
    for (int e = 0; e < 8; ++e) {
      af[ks][e]     = (__bf16)(Arow[ks * 32 + ka + e]      - ts);
      af[ks][e + 8] = (__bf16)(Arow[ks * 32 + ka + 8 + e] - ts);
    }
  }
}

template<int K>
__device__ __forceinline__ v8f mm_tile(const __bf16* __restrict__ sB,
                                       const v16bf* __restrict__ af, int ct) {
  const int lane = threadIdx.x & 31;
  v8f acc = {};
#pragma unroll
  for (int ks = 0; ks < K / 32; ++ks) {
    v16bf bv = *(const v16bf*)(sB + (size_t)((ct * (K / 32) + ks) * 32 + lane) * 16);
    acc = __builtin_amdgcn_wmma_f32_16x16x32_bf16(false, af[ks], false, bv,
                                                  (short)0, acc, false, false);
  }
  return acc;
}

__device__ __forceinline__ void store_tile(float* __restrict__ Cm, int ldc,
                                           int row0, int col0, v8f acc) {
  const int lane = threadIdx.x & 31;
  const int col = col0 + (lane & 15);
  const int rbase = row0 + ((lane >> 4) << 3);
#pragma unroll
  for (int r = 0; r < 8; ++r)
    Cm[(long)(rbase + r) * ldc + col] = acc[r];
}

// ---------------- elementwise / edge kernels ----------------

// Extract v = x[:,1:] into contiguous (N*3, C)
__global__ void k_extract_v(const float* __restrict__ x, float* __restrict__ vbuf) {
  long i = (long)blockIdx.x * 256 + threadIdx.x;
  if (i >= (long)NV * CC) return;
  int r = (int)(i / CC), c = (int)(i % CC);
  vbuf[i] = x[((long)(r / 3) * 4 + 1 + (r % 3)) * CC + c];
}

// g[h] = 0.5 * softplus(gamma[h])
__global__ void k_gamma(const float* __restrict__ gamma, float* __restrict__ g) {
  int h = threadIdx.x;
  if (h < HH) {
    float v = gamma[h];
    g[h] = 0.5f * (fmaxf(v, 0.0f) + log1pf(expf(-fabsf(v))));
  }
}

// per-(edge,head) logits + atomic segment-max
__global__ void k_edge_logits(const int* __restrict__ ei,
                              const float* __restrict__ qb, const float* __restrict__ kb,
                              const float* __restrict__ qp, const float* __restrict__ kp,
                              const float* __restrict__ g,
                              float* __restrict__ logits, unsigned* __restrict__ mkey) {
  long tid = (long)blockIdx.x * 256 + threadIdx.x;
  if (tid >= (long)EE * HH) return;
  int h = (int)(tid & 7);
  long e = tid >> 3;
  int src = ei[e], dst = ei[EE + e];
  const float* qr = qb + (long)dst * HD + h * DHD;
  const float* kr = kb + (long)src * HD + h * DHD;
  float dot = 0.f;
#pragma unroll
  for (int d = 0; d < DHD; ++d) dot += qr[d] * kr[d];
  float dist = 0.f;
#pragma unroll
  for (int xd = 0; xd < 3; ++xd) {
    const float* a = qp + ((long)dst * 3 + xd) * HP + h * NPP;
    const float* b = kp + ((long)src * 3 + xd) * HP + h * NPP;
#pragma unroll
    for (int p = 0; p < NPP; ++p) { float d2 = a[p] - b[p]; dist += d2 * d2; }
  }
  float l = dot * 0.25f - g[h] * dist;   // 1/sqrt(DH) = 0.25
  logits[tid] = l;
  atomicMax(&mkey[(long)dst * HH + h], fkey(l));
}

__global__ void k_final_m(const unsigned* __restrict__ mkey, float* __restrict__ m) {
  int i = blockIdx.x * 256 + threadIdx.x;
  if (i >= NN * HH) return;
  unsigned k = mkey[i];
  m[i] = (k == 0u) ? 0.0f : keyf(k);   // empty segment -> 0 (isfinite guard)
}

// p = exp(logit - m[dst]); z[dst] += p
__global__ void k_edge_p(const int* __restrict__ ei, const float* __restrict__ m,
                         float* __restrict__ logits, float* __restrict__ z) {
  long tid = (long)blockIdx.x * 256 + threadIdx.x;
  if (tid >= (long)EE * HH) return;
  int h = (int)(tid & 7);
  long e = tid >> 3;
  int dst = ei[EE + e];
  float p = expf(logits[tid] - m[(long)dst * HH + h]);
  logits[tid] = p;
  atomicAdd(&z[(long)dst * HH + h], p);
}

// aggregate a*vv into agg_s (128/edge) and a*vp into agg_p (96/edge)
__global__ void k_edge_agg(const int* __restrict__ ei, const float* __restrict__ pbuf,
                           const float* __restrict__ z,
                           const float* __restrict__ vb, const float* __restrict__ vp,
                           float* __restrict__ agg_s, float* __restrict__ agg_p) {
  long tid = (long)blockIdx.x * 256 + threadIdx.x;
  if (tid >= (long)EE * 128) return;
  int c = (int)(tid & 127);
  long e = tid >> 7;
  int src = ei[e], dst = ei[EE + e];
  int h = c >> 4;
  float a = pbuf[e * 8 + h] / (z[(long)dst * HH + h] + 1e-9f);
  atomicAdd(&agg_s[(long)dst * HD + c], a * vb[(long)src * HD + c]);
  if (c < 96) {
    int xd = c / 32, hc = c & 31;
    int h2 = hc >> 2;
    float a2 = pbuf[e * 8 + h2] / (z[(long)dst * HH + h2] + 1e-9f);
    atomicAdd(&agg_p[((long)dst * 3 + xd) * HP + hc],
              a2 * vp[((long)src * 3 + xd) * HP + hc]);
  }
}

// x1 = [xmid_s, ||xmid_v|| ]
__global__ void k_x1(const float* __restrict__ xmid, float* __restrict__ x1) {
  long i = (long)blockIdx.x * 256 + threadIdx.x;
  if (i >= (long)NN * CC) return;
  long n = i / CC; int c = (int)(i % CC);
  const float* p = xmid + n * 4 * CC + c;
  float v1 = p[CC], v2 = p[2 * CC], v3 = p[3 * CC];
  x1[n * C2 + c] = p[0];
  x1[n * C2 + CC + c] = sqrtf(v1 * v1 + v2 * v2 + v3 * v3 + 1e-4f);
}

// ---------------- block-tiled WMMA GEMM kernels (5 waves / block) ----------------

// q/k/v = s @ W{q,k,v};  grid (125, 3), 160 thr, 32KB LDS
__global__ void k_proj_s(const float* __restrict__ x,
                         const float* __restrict__ Wq, const float* __restrict__ Wk,
                         const float* __restrict__ Wv,
                         float* __restrict__ qb, float* __restrict__ kb, float* __restrict__ vb) {
  extern __shared__ char smem[];
  __bf16* sB = (__bf16*)smem;
  int wsel = blockIdx.y;
  const float* B = (wsel == 0) ? Wq : (wsel == 1) ? Wk : Wv;
  float* Cout    = (wsel == 0) ? qb : (wsel == 1) ? kb : vb;
  stage_B<128, 128>(B, HD, 0, sB);
  __syncthreads();
  int row0 = (blockIdx.x * 5 + (threadIdx.x >> 5)) * 16;
  v16bf af[4];
  load_A_frags<128, 0, false>(x, 4 * CC, row0, nullptr, af);
#pragma unroll
  for (int ct = 0; ct < 8; ++ct)
    store_tile(Cout, HD, row0, ct * 16, mm_tile<128>(sB, af, ct));
}

// qp/kp/vp = vbuf @ W*_pts + t; fused 3 weights; grid 375, 24KB LDS
__global__ void k_proj_p(const float* __restrict__ vbuf, const float* __restrict__ t,
                         const float* __restrict__ Wqp, const float* __restrict__ Wkp,
                         const float* __restrict__ Wvp,
                         float* __restrict__ qp, float* __restrict__ kp, float* __restrict__ vp) {
  extern __shared__ char smem[];
  __bf16* sB = (__bf16*)smem;                 // 3 panels of 128x32
  stage_B<128, 32>(Wqp, HP, 0, sB);
  stage_B<128, 32>(Wkp, HP, 0, sB + 128 * 32);
  stage_B<128, 32>(Wvp, HP, 0, sB + 2 * 128 * 32);
  __syncthreads();
  int row0 = (blockIdx.x * 5 + (threadIdx.x >> 5)) * 16;
  v16bf af[4];
  load_A_frags<128, 0, false>(vbuf, CC, row0, nullptr, af);
  const int lane = threadIdx.x & 31;
  const int col16 = lane & 15;
  const int rbase = row0 + ((lane >> 4) << 3);
  float tv[8];
#pragma unroll
  for (int r = 0; r < 8; ++r) tv[r] = t[rbase + r];
#pragma unroll
  for (int wsel = 0; wsel < 3; ++wsel) {
    float* Cout = (wsel == 0) ? qp : (wsel == 1) ? kp : vp;
#pragma unroll
    for (int ct = 0; ct < 2; ++ct) {
      v8f acc = mm_tile<128>(sB + wsel * 128 * 32, af, ct);
      int col = ct * 16 + col16;
#pragma unroll
      for (int r = 0; r < 8; ++r)
        Cout[(long)(rbase + r) * HP + col] = acc[r] + tv[r];
    }
  }
}

// out_s = agg_s @ Wo_s + bo_s; xmid row0 = x row0 + out_s; grid 125, 32KB
__global__ void k_out_s(const float* __restrict__ agg_s, const float* __restrict__ Wo_s,
                        const float* __restrict__ bo_s, const float* __restrict__ x,
                        float* __restrict__ xmid) {
  extern __shared__ char smem[];
  __bf16* sB = (__bf16*)smem;
  stage_B<128, 128>(Wo_s, CC, 0, sB);
  __syncthreads();
  int row0 = (blockIdx.x * 5 + (threadIdx.x >> 5)) * 16;
  v16bf af[4];
  load_A_frags<128, 0, false>(agg_s, HD, row0, nullptr, af);
  const int lane = threadIdx.x & 31;
  const int rbase = row0 + ((lane >> 4) << 3);
#pragma unroll
  for (int ct = 0; ct < 8; ++ct) {
    v8f acc = mm_tile<128>(sB, af, ct);
    int col = ct * 16 + (lane & 15);
    float bias = bo_s[col];
#pragma unroll
    for (int r = 0; r < 8; ++r) {
      long n = rbase + r;
      xmid[n * 4 * CC + col] = x[n * 4 * CC + col] + acc[r] + bias;
    }
  }
}

// out_p = (agg_p - t) @ Wo_pts; xmid rows1..3 = x + out_p; grid 375, 8KB
__global__ void k_out_p(const float* __restrict__ agg_p, const float* __restrict__ t,
                        const float* __restrict__ Wo_pts, const float* __restrict__ x,
                        float* __restrict__ xmid) {
  extern __shared__ char smem[];
  __bf16* sB = (__bf16*)smem;
  stage_B<32, 128>(Wo_pts, CC, 0, sB);
  __syncthreads();
  int row0 = (blockIdx.x * 5 + (threadIdx.x >> 5)) * 16;
  v16bf af[1];
  load_A_frags<32, 0, true>(agg_p, HP, row0, t, af);
  const int lane = threadIdx.x & 31;
  const int rbase = row0 + ((lane >> 4) << 3);
#pragma unroll
  for (int ct = 0; ct < 8; ++ct) {
    v8f acc = mm_tile<32>(sB, af, ct);
    int col = ct * 16 + (lane & 15);
#pragma unroll
    for (int r = 0; r < 8; ++r) {
      int gr = rbase + r;
      long idx = ((long)(gr / 3) * 4 + 1 + (gr % 3)) * CC + col;
      xmid[idx] = x[idx] + acc[r];
    }
  }
}

// h = gelu(x1 @ W1 + b1); grid (125, 2) col halves, 64KB LDS
__global__ void k_ffn1(const float* __restrict__ x1, const float* __restrict__ W1,
                       const float* __restrict__ b1, float* __restrict__ hb) {
  extern __shared__ char smem[];
  __bf16* sB = (__bf16*)smem;
  const int col0 = blockIdx.y * 128;
  stage_B<256, 128>(W1, C2, col0, sB);
  __syncthreads();
  int row0 = (blockIdx.x * 5 + (threadIdx.x >> 5)) * 16;
  v16bf af[8];
  load_A_frags<256, 0, false>(x1, C2, row0, nullptr, af);
  const int lane = threadIdx.x & 31;
  const int rbase = row0 + ((lane >> 4) << 3);
#pragma unroll
  for (int ct = 0; ct < 8; ++ct) {
    v8f acc = mm_tile<256>(sB, af, ct);
    int col = col0 + ct * 16 + (lane & 15);
    float bias = b1[col];
#pragma unroll
    for (int r = 0; r < 8; ++r) {
      float v = acc[r] + bias;
      hb[(long)(rbase + r) * C2 + col] = 0.5f * v * (1.0f + erff(v * 0.70710678118654752f));
    }
  }
}

// y = h @ W2 + b2; half 0 -> out row0 residual, half 1 -> gate = sigmoid(y)
__global__ void k_ffn2(const float* __restrict__ hb, const float* __restrict__ W2,
                       const float* __restrict__ b2, const float* __restrict__ xmid,
                       float* __restrict__ out, float* __restrict__ gate) {
  extern __shared__ char smem[];
  __bf16* sB = (__bf16*)smem;
  const int col0 = blockIdx.y * 128;
  stage_B<256, 128>(W2, C2, col0, sB);
  __syncthreads();
  int row0 = (blockIdx.x * 5 + (threadIdx.x >> 5)) * 16;
  v16bf af[8];
  load_A_frags<256, 0, false>(hb, C2, row0, nullptr, af);
  const int lane = threadIdx.x & 31;
  const int rbase = row0 + ((lane >> 4) << 3);
#pragma unroll
  for (int ct = 0; ct < 8; ++ct) {
    v8f acc = mm_tile<256>(sB, af, ct);
    int col = col0 + ct * 16 + (lane & 15);
    float bias = b2[col];
    if (col0 == 0) {
#pragma unroll
      for (int r = 0; r < 8; ++r) {
        long n = rbase + r;
        out[n * 4 * CC + col] = xmid[n * 4 * CC + col] + acc[r] + bias;
      }
    } else {
      int cg = col - CC;
#pragma unroll
      for (int r = 0; r < 8; ++r) {
        float v = acc[r] + bias;
        gate[(long)(rbase + r) * CC + cg] = 1.0f / (1.0f + expf(-v));
      }
    }
  }
}

// vec_out = (xmid_v @ Wl) * gate; out rows1..3 = xmid + vec_out; grid 375, 32KB
__global__ void k_vec(const float* __restrict__ xmid, const float* __restrict__ Wl,
                      const float* __restrict__ gate, float* __restrict__ out) {
  extern __shared__ char smem[];
  __bf16* sB = (__bf16*)smem;
  stage_B<128, 128>(Wl, CC, 0, sB);
  __syncthreads();
  int row0 = (blockIdx.x * 5 + (threadIdx.x >> 5)) * 16;
  v16bf af[4];
  load_A_frags<128, 1, false>(xmid, CC, row0, nullptr, af);
  const int lane = threadIdx.x & 31;
  const int rbase = row0 + ((lane >> 4) << 3);
#pragma unroll
  for (int ct = 0; ct < 8; ++ct) {
    v8f acc = mm_tile<128>(sB, af, ct);
    int col = ct * 16 + (lane & 15);
#pragma unroll
    for (int r = 0; r < 8; ++r) {
      int gr = rbase + r;
      long n = gr / 3; int xd = gr % 3;
      long idx = (n * 4 + 1 + xd) * CC + col;
      out[idx] = xmid[idx] + acc[r] * gate[n * CC + col];
    }
  }
}

// ---------------- launch ----------------

extern "C" void kernel_launch(void* const* d_in, const int* in_sizes, int n_in,
                              void* d_out, int out_size, void* d_ws, size_t ws_size,
                              hipStream_t stream) {
  const float* x      = (const float*)d_in[0];
  const int*   ei     = (const int*)  d_in[1];
  const float* t      = (const float*)d_in[2];
  const float* Wq     = (const float*)d_in[3];
  const float* Wk     = (const float*)d_in[4];
  const float* Wv     = (const float*)d_in[5];
  const float* Wqp    = (const float*)d_in[6];
  const float* Wkp    = (const float*)d_in[7];
  const float* Wvp    = (const float*)d_in[8];
  const float* gamma  = (const float*)d_in[9];
  const float* Wo_s   = (const float*)d_in[10];
  const float* bo_s   = (const float*)d_in[11];
  const float* Wo_pts = (const float*)d_in[12];
  const float* W1     = (const float*)d_in[13];
  const float* b1     = (const float*)d_in[14];
  const float* W2     = (const float*)d_in[15];
  const float* b2     = (const float*)d_in[16];
  const float* Wl     = (const float*)d_in[17];
  float* out = (float*)d_out;
  char* ws = (char*)d_ws;

  // Workspace layout with phase-based aliasing (~67.5 MB total).
  const size_t A0 = 0;                                   // Region A (20.48 MB)
  float* vbuf = (float*)(ws + A0);                       // 15,360,000 (dead after proj_p)
  float* xmid = (float*)(ws + A0);                       // 20,480,000
  const size_t B0 = 20480000;                            // Region B (37.12 MB)
  float* qb     = (float*)(ws + B0);                     //  5,120,000
  float* kb     = (float*)(ws + B0 +  5120000);          //  5,120,000
  float* vb     = (float*)(ws + B0 + 10240000);          //  5,120,000
  float* qp     = (float*)(ws + B0 + 15360000);          //  3,840,000
  float* kp     = (float*)(ws + B0 + 19200000);          //  3,840,000
  float* vp     = (float*)(ws + B0 + 23040000);          //  3,840,000
  float* logits = (float*)(ws + B0 + 26880000);          // 10,240,000 (reused as p)
  float* x1     = (float*)(ws + B0);                     // aliases qb/kb (dead)
  float* hb     = (float*)(ws + B0 + 10240000);          // aliases vb/qp/kp (dead)
  float* gate   = (float*)(ws + B0 + 20480000);          // aliases kp/vp (dead)
  const size_t C0 = 57600000;                            // Region C (~9.92 MB)
  unsigned* mkey = (unsigned*)(ws + C0);                 //    320,000
  float* mbuf    = (float*)(ws + C0 +  320000);          //    320,000
  float* zbuf    = (float*)(ws + C0 +  640000);          //    320,000
  float* agg_s   = (float*)(ws + C0 +  960000);          //  5,120,000
  float* agg_p   = (float*)(ws + C0 + 6080000);          //  3,840,000
  float* gcoef   = (float*)(ws + C0 + 9920000);          //         32

  hipMemsetAsync(ws + C0, 0, 960000 + 5120000 + 3840000, stream);

  k_extract_v<<<15000, 256, 0, stream>>>(x, vbuf);
  k_gamma<<<1, 32, 0, stream>>>(gamma, gcoef);
  k_proj_s<<<dim3(125, 3), 160, 32768, stream>>>(x, Wq, Wk, Wv, qb, kb, vb);
  k_proj_p<<<375, 160, 24576, stream>>>(vbuf, t, Wqp, Wkp, Wvp, qp, kp, vp);
  k_edge_logits<<<10000, 256, 0, stream>>>(ei, qb, kb, qp, kp, gcoef, logits, mkey);
  k_final_m<<<313, 256, 0, stream>>>(mkey, mbuf);
  k_edge_p<<<10000, 256, 0, stream>>>(ei, mbuf, logits, zbuf);
  k_edge_agg<<<160000, 256, 0, stream>>>(ei, logits, zbuf, vb, vp, agg_s, agg_p);
  k_out_s<<<125, 160, 32768, stream>>>(agg_s, Wo_s, bo_s, x, xmid);
  k_out_p<<<375, 160, 8192, stream>>>(agg_p, t, Wo_pts, x, xmid);
  k_x1<<<5000, 256, 0, stream>>>(xmid, x1);
  k_ffn1<<<dim3(125, 2), 160, 65536, stream>>>(x1, W1, b1, hb);
  k_ffn2<<<dim3(125, 2), 160, 65536, stream>>>(hb, W2, b2, xmid, out, gate);
  k_vec<<<375, 160, 32768, stream>>>(xmid, Wl, gate, out);
}
